// MultiHeadAttention_50757923504842
// MI455X (gfx1250) — matrix operand order repaired and checked by reading, not compiled, not run on hardware
//
#include <hip/hip_runtime.h>
#include <hip/hip_bf16.h>

// ---------------------------------------------------------------------------
// MHA forward for MI455X (gfx1250, wave32, WMMA 16x16x32 bf16)
// B=2, S=2048, D=1024, H=16, HD=64.  M = B*S = 4096, N = K = 1024.
// ---------------------------------------------------------------------------

#define CB   2
#define CS   2048
#define CD   1024
#define CH   16
#define CHD  64
#define CM   (CB*CS)     // 4096
#define CN   (CH*CHD)    // 1024
#define CK   CD          // 1024

#if defined(__has_builtin)
#  if __has_builtin(__builtin_amdgcn_global_load_async_to_lds_b128)
#    define HAS_ASYNC_LDS 1
#  else
#    define HAS_ASYNC_LDS 0
#  endif
#else
#  define HAS_ASYNC_LDS 0
#endif

typedef __attribute__((ext_vector_type(16))) __bf16        bf16x16;
typedef __attribute__((ext_vector_type(8)))  float         f32x8;
typedef __attribute__((ext_vector_type(8)))  unsigned int  u32x8;

#if HAS_ASYNC_LDS
typedef __attribute__((ext_vector_type(4))) int v4i_ty;
typedef __attribute__((address_space(1))) v4i_ty as1_v4i;   // global int4
typedef __attribute__((address_space(3))) v4i_ty as3_v4i;   // LDS int4
#endif

// async global -> LDS b128 copy (GLOBAL_LOAD_ASYNC_TO_LDS_B128, ASYNCcnt)
__device__ __forceinline__ void async_copy_b128(const void* gsrc, void* ldst) {
#if HAS_ASYNC_LDS
  __builtin_amdgcn_global_load_async_to_lds_b128(
      (as1_v4i*)gsrc, (as3_v4i*)ldst, /*offset=*/0, /*cpol=*/0);
#else
  (void)gsrc; (void)ldst;
#endif
}
__device__ __forceinline__ void wait_async0() {
#if HAS_ASYNC_LDS
#  if __has_builtin(__builtin_amdgcn_s_wait_asynccnt)
  __builtin_amdgcn_s_wait_asynccnt(0);
#  else
  asm volatile("s_wait_asynccnt 0x0" ::: "memory");
#  endif
#endif
}

__device__ __forceinline__ unsigned short f2bf(float f) {
  unsigned int u = __builtin_bit_cast(unsigned int, f);
  u += 0x7fffu + ((u >> 16) & 1u);           // round-to-nearest-even
  return (unsigned short)(u >> 16);
}

__device__ __forceinline__ bf16x16 frag16(const unsigned int* r) {
  u32x8 t;
#pragma unroll
  for (int i = 0; i < 8; ++i) t[i] = r[i];
  return __builtin_bit_cast(bf16x16, t);
}

__device__ __forceinline__ f32x8 wmma_bf16(const unsigned int* a,
                                           const unsigned int* b, f32x8 c) {
  return __builtin_amdgcn_wmma_f32_16x16x32_bf16(
      /*neg_a=*/false, frag16(a), /*neg_b=*/false, frag16(b),
      /*c_mod=*/(short)0, c, /*reuse_a=*/false, /*reuse_b=*/false);
}

// interleave two row-vectors (8 bf16 each) into pair-packed dwords
__device__ __forceinline__ void interleave8(const uint4& a, const uint4& b,
                                            unsigned int* w) {
  const unsigned int* pa = (const unsigned int*)&a;
  const unsigned int* pb = (const unsigned int*)&b;
#pragma unroll
  for (int i = 0; i < 4; ++i) {
    w[2*i]   = (pa[i] & 0xffffu)  | (pb[i] << 16);
    w[2*i+1] = (pa[i] >> 16)      | (pb[i] & 0xffff0000u);
  }
}

// ---------------------------------------------------------------------------
// fp32 -> bf16 conversion, 8 elements / thread, b128 in/out
// ---------------------------------------------------------------------------
__global__ __launch_bounds__(256) void cvt_bf16(const float* __restrict__ in,
                                                unsigned short* __restrict__ out,
                                                int n) {
  int i = (blockIdx.x * 256 + threadIdx.x) * 8;
  if (i < n) {
    float4 a = *(const float4*)(in + i);
    float4 b = *(const float4*)(in + i + 4);
    uint4 o;
    o.x = f2bf(a.x) | ((unsigned int)f2bf(a.y) << 16);
    o.y = f2bf(a.z) | ((unsigned int)f2bf(a.w) << 16);
    o.z = f2bf(b.x) | ((unsigned int)f2bf(b.y) << 16);
    o.w = f2bf(b.z) | ((unsigned int)f2bf(b.w) << 16);
    *(uint4*)(out + i) = o;
  }
}

// ---------------------------------------------------------------------------
// bf16 GEMM: C[4096,1024] = A[4096,1024] x B[1024,1024] + bias
// 128x128 block tile, BK=32, 256 threads = 8 waves (2x4), 4x2 WMMA tiles/wave.
// Ping-pong LDS; A tile via async global->LDS copy, B tile via register
// interleave; next tile always in flight during the WMMAs.
// ---------------------------------------------------------------------------
__global__ __launch_bounds__(256) void gemm128(
    const unsigned short* __restrict__ Aq,   // [M,K] bf16
    const unsigned short* __restrict__ Bw,   // [K,N] bf16
    const float* __restrict__ bias,          // [N] fp32
    void* __restrict__ outp, int mode) {
  __shared__ unsigned int As[2][128 * 20];   // [row][16 dwords] stride 20
  __shared__ unsigned int Bs[2][16 * 132];   // pair-packed [k/2][n] stride 132

  constexpr bool kAsync = HAS_ASYNC_LDS != 0;

  const int tid  = threadIdx.x;
  const int wave = tid >> 5;
  const int lane = tid & 31;
  const int g    = lane >> 4;
  const int ln   = lane & 15;
  const int wr   = wave >> 2;     // 0..1
  const int wc   = wave & 3;      // 0..3
  const int m0   = blockIdx.y * 128;
  const int n0   = blockIdx.x * 128;

  const uint4* Ag4 = (const uint4*)Aq;   // row stride = 128 uint4
  const uint4* Bg4 = (const uint4*)Bw;   // row stride = 128 uint4

  const int aRow0 = tid >> 2, aSeg0 = tid & 3;
  const int aRow1 = (tid + 256) >> 2, aSeg1 = (tid + 256) & 3;
  const int bP   = tid >> 4;             // k-pair 0..15
  const int bSeg = tid & 15;             // 8-col segment

  uint4 ra0, ra1, rb0, rb1;
  auto loadA = [&](int k0) {
    ra0 = Ag4[(size_t)(m0 + aRow0) * 128 + (k0 >> 3) + aSeg0];
    ra1 = Ag4[(size_t)(m0 + aRow1) * 128 + (k0 >> 3) + aSeg1];
  };
  auto storeA = [&](int buf) {
    *(uint4*)&As[buf][aRow0 * 20 + aSeg0 * 4] = ra0;
    *(uint4*)&As[buf][aRow1 * 20 + aSeg1 * 4] = ra1;
  };
  auto issueA = [&](int k0, int buf) {   // async copy path
    async_copy_b128(&Ag4[(size_t)(m0 + aRow0) * 128 + (k0 >> 3) + aSeg0],
                    &As[buf][aRow0 * 20 + aSeg0 * 4]);
    async_copy_b128(&Ag4[(size_t)(m0 + aRow1) * 128 + (k0 >> 3) + aSeg1],
                    &As[buf][aRow1 * 20 + aSeg1 * 4]);
  };
  auto loadB = [&](int k0) {
    rb0 = Bg4[(size_t)(k0 + 2 * bP) * 128 + (n0 >> 3) + bSeg];
    rb1 = Bg4[(size_t)(k0 + 2 * bP + 1) * 128 + (n0 >> 3) + bSeg];
  };
  auto storeB = [&](int buf) {
    unsigned int w[8];
    interleave8(rb0, rb1, w);
    *(uint4*)&Bs[buf][bP * 132 + bSeg * 8]     = *(uint4*)&w[0];
    *(uint4*)&Bs[buf][bP * 132 + bSeg * 8 + 4] = *(uint4*)&w[4];
  };

  f32x8 acc[4][2];
#pragma unroll
  for (int i = 0; i < 4; ++i)
#pragma unroll
    for (int j = 0; j < 2; ++j) acc[i][j] = (f32x8)(0.0f);

  if constexpr (kAsync) { issueA(0, 0); } else { loadA(0); storeA(0); }
  loadB(0);
  storeB(0);

  for (int kt = 0; kt < CK / 32; ++kt) {
    const int cur = kt & 1;
    const bool hasNext = (kt + 1) < (CK / 32);
    wait_async0();                 // drain async copies into buf[cur]
    __syncthreads();
    if (hasNext) {                 // buf[1-cur] free to fill from here on
      if constexpr (kAsync) issueA((kt + 1) * 32, 1 - cur);
      else                  loadA((kt + 1) * 32);
      loadB((kt + 1) * 32);
    }

    // ---- A fragments (ISA 16-bit A 16x32 swizzle) -> ds_load_b128 pairs
    unsigned int a[4][8];
#pragma unroll
    for (int i = 0; i < 4; ++i) {
      int m = wr * 64 + i * 16 + ln;
#pragma unroll
      for (int v = 0; v < 8; ++v) {
        int k = ((v >> 2) << 4) + (g << 3) + ((v & 3) << 1);
        a[i][v] = As[cur][m * 20 + (k >> 1)];
      }
    }
    // ---- B fragments: pair-packed dwords, one ds_load_b32 per VGPR
    unsigned int b[2][8];
#pragma unroll
    for (int j = 0; j < 2; ++j) {
      int n = wc * 32 + j * 16 + ln;
#pragma unroll
      for (int v = 0; v < 8; ++v)
        b[j][v] = Bs[cur][(((v >> 2) << 3) + (g << 2) + (v & 3)) * 132 + n];
    }
#pragma unroll
    for (int i = 0; i < 4; ++i)
#pragma unroll
      for (int j = 0; j < 2; ++j)
        acc[i][j] = wmma_bf16(a[i], b[j], acc[i][j]);

    if (hasNext) {
      if constexpr (!kAsync) storeA(1 - cur);
      storeB(1 - cur);
    }
  }

  // ---- epilogue
#pragma unroll
  for (int i = 0; i < 4; ++i) {
#pragma unroll
    for (int j = 0; j < 2; ++j) {
      int n = n0 + wc * 32 + j * 16 + ln;
      float bv = bias[n];
#pragma unroll
      for (int v = 0; v < 8; ++v) {
        int m = m0 + wr * 64 + i * 16 + v + (g << 3);
        float val = acc[i][j][v] + bv;
        if (mode == 0) {
          int bb = m >> 11, s = m & (CS - 1);
          int h = n >> 6, hd = n & (CHD - 1);
          ((unsigned short*)outp)[((size_t)((bb * CH + h) * CS + s) << 6) + hd] =
              f2bf(val);
        } else {
          ((float*)outp)[(size_t)m * CN + n] = val;
        }
      }
    }
  }
}

// ---------------------------------------------------------------------------
// Flash attention: one (b,h) x 64 query rows per block (4 waves x 16 rows).
// 32-key chunks; K tile via async global->LDS copy, V via register interleave,
// ping-pong LDS overlapped with compute; online softmax in f32.
// ---------------------------------------------------------------------------
__global__ __launch_bounds__(128) void attn64(
    const unsigned short* __restrict__ Qh,   // [B,H,S,HD] bf16
    const unsigned short* __restrict__ Kh,
    const unsigned short* __restrict__ Vh,
    unsigned short* __restrict__ AO) {       // [B*S, H*HD] bf16
  __shared__ unsigned int Ks[2][32 * 36];    // [key][hd dwords] stride 36
  __shared__ unsigned int Vs[2][16 * 68];    // pair-packed [key/2][hd] stride 68
  __shared__ unsigned short Pw[4 * 16 * 34]; // per-wave P re-swizzle tile

  constexpr bool kAsync = HAS_ASYNC_LDS != 0;

  const int tid  = threadIdx.x;
  const int wave = tid >> 5;
  const int lane = tid & 31;
  const int g    = lane >> 4;
  const int ln   = lane & 15;

  const int bh = blockIdx.y;                           // b*H + h
  const int q0 = (gridDim.x - 1 - blockIdx.x) * 64;    // longest tiles first
  const int qrb = q0 + wave * 16;

  const size_t base_dw = (size_t)bh * CS * (CHD / 2);
  const unsigned int* Qg = (const unsigned int*)Qh + base_dw;
  const uint4* Kg4 = (const uint4*)((const unsigned int*)Kh + base_dw); // 8/row
  const uint4* Vg4 = (const uint4*)((const unsigned int*)Vh + base_dw);

  const int kRow0 = tid >> 3, kSeg0 = tid & 7;
  const int kRow1 = (tid + 128) >> 3, kSeg1 = (tid + 128) & 7;
  const int vP = tid >> 3, vSeg = tid & 7;

  uint4 rk0, rk1, rv0, rv1;
  auto loadK = [&](int k0) {
    rk0 = Kg4[(size_t)(k0 + kRow0) * 8 + kSeg0];
    rk1 = Kg4[(size_t)(k0 + kRow1) * 8 + kSeg1];
  };
  auto storeK = [&](int buf) {
    *(uint4*)&Ks[buf][kRow0 * 36 + kSeg0 * 4] = rk0;
    *(uint4*)&Ks[buf][kRow1 * 36 + kSeg1 * 4] = rk1;
  };
  auto issueK = [&](int k0, int buf) {
    async_copy_b128(&Kg4[(size_t)(k0 + kRow0) * 8 + kSeg0],
                    &Ks[buf][kRow0 * 36 + kSeg0 * 4]);
    async_copy_b128(&Kg4[(size_t)(k0 + kRow1) * 8 + kSeg1],
                    &Ks[buf][kRow1 * 36 + kSeg1 * 4]);
  };
  auto loadV = [&](int k0) {
    rv0 = Vg4[(size_t)(k0 + 2 * vP) * 8 + vSeg];
    rv1 = Vg4[(size_t)(k0 + 2 * vP + 1) * 8 + vSeg];
  };
  auto storeV = [&](int buf) {
    unsigned int w[8];
    interleave8(rv0, rv1, w);
    *(uint4*)&Vs[buf][vP * 68 + vSeg * 8]     = *(uint4*)&w[0];
    *(uint4*)&Vs[buf][vP * 68 + vSeg * 8 + 4] = *(uint4*)&w[4];
  };

  // Q fragments: two 16x32 A-frags covering HD=64
  unsigned int qa[2][8];
#pragma unroll
  for (int c = 0; c < 2; ++c)
#pragma unroll
    for (int v = 0; v < 8; ++v) {
      int k = c * 32 + ((v >> 2) << 4) + (g << 3) + ((v & 3) << 1);
      qa[c][v] = Qg[(size_t)(qrb + ln) * 32 + (k >> 1)];
    }

  f32x8 o[4];
#pragma unroll
  for (int j = 0; j < 4; ++j) o[j] = (f32x8)(0.0f);
  float mrow[8], lrow[8];
#pragma unroll
  for (int v = 0; v < 8; ++v) { mrow[v] = -1e30f; lrow[v] = 0.0f; }

  const int nch = (q0 >> 5) + 2;        // causal: keys <= q0+63
  if constexpr (kAsync) { issueK(0, 0); } else { loadK(0); storeK(0); }
  loadV(0);
  storeV(0);

  for (int ck = 0; ck < nch; ++ck) {
    const int cur = ck & 1;
    const bool hasNext = (ck + 1) < nch;
    const int k0 = ck << 5;
    wait_async0();
    __syncthreads();
    if (hasNext) {
      if constexpr (kAsync) issueK((ck + 1) << 5, 1 - cur);
      else                  loadK((ck + 1) << 5);
      loadV((ck + 1) << 5);
    }

    // ---- scores for two 16-key sub-tiles
    f32x8 s[2];
#pragma unroll
    for (int t = 0; t < 2; ++t) {
      f32x8 z = (f32x8)(0.0f);
#pragma unroll
      for (int cc = 0; cc < 2; ++cc) {
        unsigned int kb[8];
#pragma unroll
        for (int v = 0; v < 8; ++v)
          kb[v] = Ks[cur][(t * 16 + ln) * 36 + cc * 16 + ((v >> 2) << 3) + (g << 2) + (v & 3)];
        z = wmma_bf16(qa[cc], kb, z);
      }
      s[t] = z;
    }
    // ---- scale + causal mask
#pragma unroll
    for (int t = 0; t < 2; ++t)
#pragma unroll
      for (int v = 0; v < 8; ++v) {
        float val = s[t][v] * 0.125f;                 // 1/sqrt(64)
        int key = k0 + t * 16 + ln;
        int qr  = qrb + v + (g << 3);
        s[t][v] = (key > qr) ? -1e9f : val;
      }
    // ---- online softmax (16-lane xor reductions)
    float al[8];
#pragma unroll
    for (int v = 0; v < 8; ++v) {
      float x = fmaxf(s[0][v], s[1][v]);
      x = fmaxf(x, __shfl_xor(x, 1, 32));
      x = fmaxf(x, __shfl_xor(x, 2, 32));
      x = fmaxf(x, __shfl_xor(x, 4, 32));
      x = fmaxf(x, __shfl_xor(x, 8, 32));
      float mn = fmaxf(mrow[v], x);
      al[v] = __expf(mrow[v] - mn);
      mrow[v] = mn;
    }
#pragma unroll
    for (int t = 0; t < 2; ++t)
#pragma unroll
      for (int v = 0; v < 8; ++v) s[t][v] = __expf(s[t][v] - mrow[v]);
#pragma unroll
    for (int v = 0; v < 8; ++v) {
      float x = s[0][v] + s[1][v];
      x += __shfl_xor(x, 1, 32);
      x += __shfl_xor(x, 2, 32);
      x += __shfl_xor(x, 4, 32);
      x += __shfl_xor(x, 8, 32);
      lrow[v] = lrow[v] * al[v] + x;
    }
    // ---- P (C-layout) -> LDS -> A-layout bf16  (wave-private, DS in-order)
    unsigned short* P = &Pw[wave * 16 * 34];
#pragma unroll
    for (int t = 0; t < 2; ++t)
#pragma unroll
      for (int v = 0; v < 8; ++v)
        P[(v + (g << 3)) * 34 + t * 16 + ln] = f2bf(s[t][v]);
    unsigned int pa[8];
#pragma unroll
    for (int v = 0; v < 8; ++v) {
      int k = ((v >> 2) << 4) + (g << 3) + ((v & 3) << 1);
      pa[v] = *(const unsigned int*)&P[ln * 34 + k];
    }
    // ---- O = diag(alpha)*O + P @ V  (running accumulator as WMMA C)
#pragma unroll
    for (int j = 0; j < 4; ++j) {
#pragma unroll
      for (int v = 0; v < 8; ++v) o[j][v] *= al[v];
      unsigned int vb[8];
#pragma unroll
      for (int v = 0; v < 8; ++v)
        vb[v] = Vs[cur][(((v >> 2) << 3) + (g << 2) + (v & 3)) * 68 + j * 16 + ln];
      o[j] = wmma_bf16(pa, vb, o[j]);
    }

    if (hasNext) {
      if constexpr (!kAsync) storeK(1 - cur);
      storeV(1 - cur);
    }
  }

  // ---- finalize: O /= l, write bf16 [B*S, H*HD]
  const int b = bh >> 4, h = bh & (CH - 1);
#pragma unroll
  for (int v = 0; v < 8; ++v) {
    float inv = 1.0f / lrow[v];
    int srow = qrb + v + (g << 3);
    size_t rbase = (size_t)(b * CS + srow) * CN + h * CHD;
#pragma unroll
    for (int j = 0; j < 4; ++j)
      AO[rbase + j * 16 + ln] = f2bf(o[j][v] * inv);
  }
}

// ---------------------------------------------------------------------------
// launcher
// ---------------------------------------------------------------------------
extern "C" void kernel_launch(void* const* d_in, const int* in_sizes, int n_in,
                              void* d_out, int out_size, void* d_ws,
                              size_t ws_size, hipStream_t stream) {
  const float* x  = (const float*)d_in[0];
  // d_in[1] = causal mask (recomputed analytically)
  const float* wq = (const float*)d_in[2];
  const float* bq = (const float*)d_in[3];
  const float* wk = (const float*)d_in[4];
  const float* bk = (const float*)d_in[5];
  const float* wv = (const float*)d_in[6];
  const float* bv = (const float*)d_in[7];
  const float* wo = (const float*)d_in[8];
  const float* bo = (const float*)d_in[9];
  float* out = (float*)d_out;

  // workspace layout (bf16 elements)
  unsigned short* xb  = (unsigned short*)d_ws;       // [4096,1024]
  unsigned short* wqb = xb  + (size_t)CM * CK;
  unsigned short* wkb = wqb + (size_t)CK * CN;
  unsigned short* wvb = wkb + (size_t)CK * CN;
  unsigned short* wob = wvb + (size_t)CK * CN;
  unsigned short* qh  = wob + (size_t)CK * CN;       // [B,H,S,HD]
  unsigned short* kh  = qh  + (size_t)CM * CN;
  unsigned short* vh  = kh  + (size_t)CM * CN;
  unsigned short* ao  = vh  + (size_t)CM * CN;       // [4096,1024]

  cvt_bf16<<<(CM * CK) / 2048, 256, 0, stream>>>(x, xb, CM * CK);
  cvt_bf16<<<(CK * CN) / 2048, 256, 0, stream>>>(wq, wqb, CK * CN);
  cvt_bf16<<<(CK * CN) / 2048, 256, 0, stream>>>(wk, wkb, CK * CN);
  cvt_bf16<<<(CK * CN) / 2048, 256, 0, stream>>>(wv, wvb, CK * CN);
  cvt_bf16<<<(CK * CN) / 2048, 256, 0, stream>>>(wo, wob, CK * CN);

  dim3 gg(CN / 128, CM / 128);
  gemm128<<<gg, 256, 0, stream>>>(xb, wqb, bq, qh, 0);
  gemm128<<<gg, 256, 0, stream>>>(xb, wkb, bk, kh, 0);
  gemm128<<<gg, 256, 0, stream>>>(xb, wvb, bv, vh, 0);

  attn64<<<dim3(CS / 64, CB * CH), 128, 0, stream>>>(qh, kh, vh, ao);

  gemm128<<<gg, 256, 0, stream>>>(ao, wob, bo, out, 1);
}
